// ChiralGNN_42253888258097
// MI455X (gfx1250) — matrix operand order repaired and checked by reading, not compiled, not run on hardware
//
#include <hip/hip_runtime.h>

// ---------------------------------------------------------------------------
// Problem constants (from reference)
// ---------------------------------------------------------------------------
constexpr int NNODE = 50000;
constexpr int NEDGE = 200000;
constexpr int NGRAPH = 2048;
constexpr int DD = 256;   // feature dim
constexpr int HH = 512;   // hidden dim
constexpr int LLAYER = 12;

// GEMM tiling
constexpr int BM = 128;
constexpr int BN = 128;
constexpr int BK = 32;
constexpr int LDSP = 40;  // LDS row pitch (ushorts) = 80B; TDM pad reproduces this

typedef unsigned short bhalf;  // bf16 bit pattern
typedef __attribute__((ext_vector_type(16))) __bf16 v16bf;
typedef __attribute__((ext_vector_type(8)))  float  v8f;
typedef __attribute__((ext_vector_type(4)))  unsigned int su4;
typedef __attribute__((ext_vector_type(8)))  unsigned int su8;

union Frag { uint4 q[2]; v16bf v; };

__device__ __forceinline__ bhalf f2bf(float f) {
  unsigned int u; __builtin_memcpy(&u, &f, 4);
  unsigned int r = u + 0x7FFFu + ((u >> 16) & 1u);   // round-to-nearest-even
  return (bhalf)(r >> 16);
}

// ---------------------------------------------------------------------------
// Tensor Data Mover: 2D tile (tile_w x tile_h elements, 2B each) from global
// row-major [rows x row_stride] into LDS with hardware padding:
//   pad_interval=3 (every 16 DWORDs = one 32-elem bf16 row)
//   pad_amount  =3 (4 DWORDs = 16B)  -> LDS pitch 80B = LDSP ushorts.
// D# per CDNA5 ISA 8.3/8.4 (group0: count|lds|global|type=2; group1: dims).
// EXEC is ignored by tensor ops; guard by wave so each issues once.
// ---------------------------------------------------------------------------
__device__ __forceinline__ void tdm_load_tile(
    unsigned lds_byte_addr, const void* gaddr,
    unsigned tensor_w,   // valid elements per row from tile start (OOB -> 0)
    unsigned tensor_h,   // valid rows from tile start (OOB -> 0)
    unsigned tile_w, unsigned tile_h,
    unsigned row_stride) // elements
{
  const unsigned long long ga = (unsigned long long)gaddr;
  su4 g0;
  g0[0] = 1u;                                   // count=1, user descriptor
  g0[1] = lds_byte_addr;                        // lds_addr (bytes)
  g0[2] = (unsigned)ga;                         // global_addr[31:0]
  g0[3] = (unsigned)((ga >> 32) & 0x1FFFFFFu)   // global_addr[56:32]
        | (2u << 30);                           // type = 2 ("image")
  su8 g1;
  g1[0] = (1u << 16)                            // data_size = 1 -> 2 bytes
        | (1u << 20)                            // pad_enable
        | (3u << 22)                            // pad_interval: 16 DWORDs
        | (3u << 25);                           // pad_amount: 4 DWORDs
  g1[1] = (tensor_w & 0xFFFFu) << 16;           // tensor_dim0[15:0]
  g1[2] = (tensor_w >> 16) | ((tensor_h & 0xFFFFu) << 16);  // dim0 hi | dim1 lo
  g1[3] = (tensor_h >> 16) | (tile_w << 16);    // dim1 hi | tile_dim0
  g1[4] = tile_h & 0xFFFFu;                     // tile_dim1 (tile_dim2 = 0)
  g1[5] = row_stride;                           // tensor_dim0_stride[31:0]
  g1[6] = 0u;                                   // stride hi | dim1_stride lo
  g1[7] = 0u;
  asm volatile("tensor_load_to_lds %0, %1" :: "s"(g0), "s"(g1) : "memory");
}

// ---------------------------------------------------------------------------
// WMMA GEMM: C[M,Nn] = epilogue(A[M,K]_bf16 @ Wt[Nn,K]_bf16^T + bias)
//   MODE 0: outB = bf16(relu(v))
//   MODE 1: outF = relu(v); outB = bf16(outF)
//   MODE 2: outF += relu(v) (residual, in place); outB = bf16(outF)
//   MODE 3: outF += v       (position head accumulate, no relu, no outB)
// K multiple of 32. 256 threads = 8 wave32s; wave grid 4(M) x 2(N);
// wave tile 32x64 = 8 v_wmma_f32_16x16x32_bf16 per K-step.
// Tiles stream into double-buffered LDS via the Tensor Data Mover.
// ---------------------------------------------------------------------------
template<int MODE>
__global__ __launch_bounds__(256)
void gemm_bf16_kernel(const bhalf* __restrict__ A, const bhalf* __restrict__ Wt,
                      const float* __restrict__ bias,
                      bhalf* __restrict__ outB, float* __restrict__ outF,
                      int M, int Nn, int K)
{
  __shared__ bhalf As[2][BM * LDSP];
  __shared__ bhalf Bs[2][BN * LDSP];

  const int tid  = threadIdx.x;
  const int lane = tid & 31;
  const int wave = tid >> 5;
  const int wm   = wave & 3;   // 0..3 -> M subtile of 32
  const int wn   = wave >> 2;  // 0..1 -> N subtile of 64
  const int hf   = lane >> 4;  // half-wave select
  const int lr   = lane & 15;
  const int m0   = blockIdx.x * BM;
  const int n0   = blockIdx.y * BN;

  v8f acc[2][4];
#pragma unroll
  for (int i = 0; i < 2; ++i)
#pragma unroll
    for (int j = 0; j < 4; ++j)
      acc[i][j] = v8f{0.f,0.f,0.f,0.f,0.f,0.f,0.f,0.f};

  const int KT = K / BK;

  // wave 0 DMAs the A tile, wave 1 DMAs the B tile (one TDM op each)
  auto issue_tdm = [&](int kt, int buf) {
    const int kbase = kt * BK;
    if (wave == 0) {
      tdm_load_tile((unsigned)(uintptr_t)&As[buf][0],
                    A + (size_t)m0 * K + kbase,
                    (unsigned)(K - kbase), (unsigned)(M - m0),
                    BK, BM, (unsigned)K);
    } else if (wave == 1) {
      tdm_load_tile((unsigned)(uintptr_t)&Bs[buf][0],
                    Wt + (size_t)n0 * K + kbase,
                    (unsigned)(K - kbase), (unsigned)(Nn - n0),
                    BK, BN, (unsigned)K);
    }
  };
  auto drain_tdm = [&]() {
    if (wave < 2) __builtin_amdgcn_s_wait_tensorcnt(0);
    __syncthreads();
  };

  issue_tdm(0, 0);
  drain_tdm();

  for (int kt = 0; kt < KT; ++kt) {
    const int buf = kt & 1;
    if (kt + 1 < KT) issue_tdm(kt + 1, buf ^ 1);  // stream next tile into other buffer

    Frag afr[2], bfr[4];
#pragma unroll
    for (int mt = 0; mt < 2; ++mt) {
      // A 16x32 bf16 fragment: low lanes K 0-7 & 16-23, high lanes +8
      const bhalf* p = &As[buf][(wm * 32 + mt * 16 + lr) * LDSP + hf * 8];
      afr[mt].q[0] = *(const uint4*)p;
      afr[mt].q[1] = *(const uint4*)(p + 16);
    }
#pragma unroll
    for (int nt = 0; nt < 4; ++nt) {
      // B 32x16 bf16 fragment: low lanes K 0-7 & 16-23, high lanes +8
      const bhalf* p = &Bs[buf][(wn * 64 + nt * 16 + lr) * LDSP + hf * 8];
      bfr[nt].q[0] = *(const uint4*)p;
      bfr[nt].q[1] = *(const uint4*)(p + 16);
    }
#pragma unroll
    for (int mt = 0; mt < 2; ++mt)
#pragma unroll
      for (int nt = 0; nt < 4; ++nt)
        acc[mt][nt] = __builtin_amdgcn_wmma_f32_16x16x32_bf16(
            false, afr[mt].v, false, bfr[nt].v, (short)0, acc[mt][nt],
            false, false);

    drain_tdm();
  }

  // epilogue: C layout lane lr = N, acc element i -> M = hf*8 + i
#pragma unroll
  for (int mt = 0; mt < 2; ++mt) {
#pragma unroll
    for (int nt = 0; nt < 4; ++nt) {
      const int n = n0 + wn * 64 + nt * 16 + lr;
      if (n >= Nn) continue;
      const float bv = bias ? bias[n] : 0.0f;
#pragma unroll
      for (int i = 0; i < 8; ++i) {
        const int m = m0 + wm * 32 + mt * 16 + hf * 8 + i;
        if (m >= M) continue;
        const float v = acc[mt][nt][i] + bv;
        const size_t idx = (size_t)m * Nn + n;
        if (MODE == 0) {
          outB[idx] = f2bf(fmaxf(v, 0.0f));
        } else if (MODE == 1) {
          const float r = fmaxf(v, 0.0f);
          outF[idx] = r; outB[idx] = f2bf(r);
        } else if (MODE == 2) {
          const float r = outF[idx] + fmaxf(v, 0.0f);
          outF[idx] = r; outB[idx] = f2bf(r);
        } else {
          outF[idx] += v;
        }
      }
    }
  }
}

// ---------------------------------------------------------------------------
// Glue kernels
// ---------------------------------------------------------------------------
__global__ void zero_f32_kernel(float* p, long n) {
  long i = (long)blockIdx.x * blockDim.x + threadIdx.x;
  const long stride = (long)gridDim.x * blockDim.x;
  for (; i < n; i += stride) p[i] = 0.0f;
}

// f32 rows [M,Kin] -> bf16 rows [M,Kp] (K zero-padded to multiple of 32)
__global__ void pack_rows_kernel(const float* __restrict__ s, bhalf* __restrict__ d,
                                 int M, int Kin, int Kp) {
  const long tot = (long)M * Kp;
  long i = (long)blockIdx.x * blockDim.x + threadIdx.x;
  const long stride = (long)gridDim.x * blockDim.x;
  for (; i < tot; i += stride) {
    const int m = (int)(i / Kp), k = (int)(i % Kp);
    d[i] = (k < Kin) ? f2bf(s[(size_t)m * Kin + k]) : (bhalf)0;
  }
}

// f32 W[K,Nn] row-major -> bf16 Wt[Nn,Kp] (transposed, K zero-padded)
__global__ void pack_wt_kernel(const float* __restrict__ W, bhalf* __restrict__ Wt,
                               int K, int Nn, int Kp) {
  const long tot = (long)Nn * Kp;
  long i = (long)blockIdx.x * blockDim.x + threadIdx.x;
  const long stride = (long)gridDim.x * blockDim.x;
  for (; i < tot; i += stride) {
    const int n = (int)(i / Kp), k = (int)(i % Kp);
    Wt[i] = (k < K) ? f2bf(W[(size_t)k * Nn + n]) : (bhalf)0;
  }
}

__global__ void init_u_kernel(const float* __restrict__ u0,
                              float* __restrict__ uF, bhalf* __restrict__ uB) {
  const int g = blockIdx.x, t = threadIdx.x;
  const float v = u0[t];
  uF[(size_t)g * DD + t] = v;
  uB[(size_t)g * DD + t] = f2bf(v);
}

// e_in[ed] = concat(e[ed], x[src], x[dst], u[batch(src)]); 16B per lane.
// 128 threads: seg = t>>5 (4 segments of 256 bf16), c = t&31 (32 x uint4).
__global__ void build_e_in_kernel(const bhalf* __restrict__ eB, const bhalf* __restrict__ xB,
                                  const bhalf* __restrict__ uB,
                                  const int* __restrict__ srcI, const int* __restrict__ dstI,
                                  const int* __restrict__ nb, bhalf* __restrict__ eIn) {
  const int ed = blockIdx.x, t = threadIdx.x;
  const int s  = srcI[ed], d2 = dstI[ed];
  const int b  = nb[s];
  __builtin_prefetch(xB + (size_t)s  * DD, 0, 0);   // -> global_prefetch_b8
  __builtin_prefetch(xB + (size_t)d2 * DD, 0, 0);
  const int seg = t >> 5, c = t & 31;
  const uint4* src;
  if      (seg == 0) src = (const uint4*)(eB + (size_t)ed * DD);
  else if (seg == 1) src = (const uint4*)(xB + (size_t)s  * DD);
  else if (seg == 2) src = (const uint4*)(xB + (size_t)d2 * DD);
  else               src = (const uint4*)(uB + (size_t)b  * DD);
  ((uint4*)(eIn + (size_t)ed * 4 * DD))[seg * 32 + c] = src[c];
}

__global__ void scatter_e_kernel(const float* __restrict__ eF,
                                 const int* __restrict__ srcI, const int* __restrict__ dstI,
                                 float* __restrict__ sentF, float* __restrict__ recvF) {
  const int ed = blockIdx.x, t = threadIdx.x;
  const float v = eF[(size_t)ed * DD + t];
  atomicAdd(&sentF[(size_t)srcI[ed] * DD + t], v);
  atomicAdd(&recvF[(size_t)dstI[ed] * DD + t], v);
}

// n_in = concat(x, sent, recv, u[batch]); 128 threads, 16B per lane.
__global__ void build_n_in_kernel(const bhalf* __restrict__ xB,
                                  const float* __restrict__ sentF, const float* __restrict__ recvF,
                                  const bhalf* __restrict__ uB, const int* __restrict__ nb,
                                  bhalf* __restrict__ nIn) {
  const int n = blockIdx.x, t = threadIdx.x;
  const int seg = t >> 5, c = t & 31;
  uint4* out = (uint4*)(nIn + (size_t)n * 4 * DD);
  if (seg == 0) {
    out[c] = ((const uint4*)(xB + (size_t)n * DD))[c];
  } else if (seg == 3) {
    out[96 + c] = ((const uint4*)(uB + (size_t)nb[n] * DD))[c];
  } else {
    const float* sf = ((seg == 1) ? sentF : recvF) + (size_t)n * DD + c * 8;
    bhalf tmp[8];
#pragma unroll
    for (int i = 0; i < 8; ++i) tmp[i] = f2bf(sf[i]);
    uint4 v; __builtin_memcpy(&v, tmp, 16);
    out[seg * 32 + c] = v;
  }
}

__global__ void scatter_gn_kernel(const float* __restrict__ xF, const int* __restrict__ nb,
                                  float* __restrict__ gnF) {
  const int n = blockIdx.x, t = threadIdx.x;
  atomicAdd(&gnF[(size_t)nb[n] * DD + t], xF[(size_t)n * DD + t]);
}

__global__ void scatter_ge_kernel(const float* __restrict__ eF, const int* __restrict__ srcI,
                                  const int* __restrict__ nb, float* __restrict__ geF) {
  const int ed = blockIdx.x, t = threadIdx.x;
  atomicAdd(&geF[(size_t)nb[srcI[ed]] * DD + t], eF[(size_t)ed * DD + t]);
}

// g_in = concat(u, sum_x, sum_e); 96 threads, 16B per lane.
__global__ void build_g_in_kernel(const bhalf* __restrict__ uB,
                                  const float* __restrict__ gnF, const float* __restrict__ geF,
                                  bhalf* __restrict__ gIn) {
  const int g = blockIdx.x, t = threadIdx.x;
  const int seg = t >> 5, c = t & 31;
  uint4* out = (uint4*)(gIn + (size_t)g * 3 * DD);
  if (seg == 0) {
    out[c] = ((const uint4*)(uB + (size_t)g * DD))[c];
  } else {
    const float* sf = ((seg == 1) ? gnF : geF) + (size_t)g * DD + c * 8;
    bhalf tmp[8];
#pragma unroll
    for (int i = 0; i < 8; ++i) tmp[i] = f2bf(sf[i]);
    uint4 v; __builtin_memcpy(&v, tmp, 16);
    out[seg * 32 + c] = v;
  }
}

// ---------------------------------------------------------------------------
// Host orchestration
// ---------------------------------------------------------------------------
extern "C" void kernel_launch(void* const* d_in, const int* in_sizes, int n_in,
                              void* d_out, int out_size, void* d_ws, size_t ws_size,
                              hipStream_t stream)
{
  (void)in_sizes; (void)n_in; (void)out_size; (void)ws_size;

  const float* x_raw    = (const float*)d_in[0];
  const float* edge_raw = (const float*)d_in[1];
  const float* u0       = (const float*)d_in[2];
  const float *enW1 = (const float*)d_in[3],  *enb1 = (const float*)d_in[4];
  const float *enW2 = (const float*)d_in[5],  *enb2 = (const float*)d_in[6];
  const float *enW3 = (const float*)d_in[7],  *enb3 = (const float*)d_in[8];
  const float *eeW1 = (const float*)d_in[9],  *eeb1 = (const float*)d_in[10];
  const float *eeW2 = (const float*)d_in[11], *eeb2 = (const float*)d_in[12];
  const float *eeW3 = (const float*)d_in[13], *eeb3 = (const float*)d_in[14];
  const float *eW1  = (const float*)d_in[15], *eb1  = (const float*)d_in[16];
  const float *eW2  = (const float*)d_in[17], *eb2  = (const float*)d_in[18];
  const float *eW3  = (const float*)d_in[19], *eb3  = (const float*)d_in[20];
  const float *nW1  = (const float*)d_in[21], *nb1  = (const float*)d_in[22];
  const float *nW2  = (const float*)d_in[23], *nb2  = (const float*)d_in[24];
  const float *nW3  = (const float*)d_in[25], *nb3  = (const float*)d_in[26];
  const float *gW1  = (const float*)d_in[27], *gb1  = (const float*)d_in[28];
  const float *gW2  = (const float*)d_in[29], *gb2  = (const float*)d_in[30];
  const float *gW3  = (const float*)d_in[31], *gb3  = (const float*)d_in[32];
  const float *pW1  = (const float*)d_in[33], *pb1  = (const float*)d_in[34];
  const float *pW2  = (const float*)d_in[35], *pb2  = (const float*)d_in[36];
  const int* edge_index = (const int*)d_in[37];
  const int* node_batch = (const int*)d_in[38];
  const int* srcI = edge_index;
  const int* dstI = edge_index + NEDGE;
  float* pos = (float*)d_out;

  // workspace allocator
  size_t off = 0;
  char* base = (char*)d_ws;
  auto alloc = [&](size_t bytes) -> void* {
    void* p = base + off;
    off += (bytes + 255) & ~(size_t)255;
    return p;
  };
  auto aB = [&](size_t e) -> bhalf* { return (bhalf*)alloc(e * sizeof(bhalf)); };
  auto aF = [&](size_t e) -> float* { return (float*)alloc(e * sizeof(float)); };

  // packed bf16 weights (N-major, K padded to 32)
  bhalf* enW1t = aB((size_t)512 * 192);
  bhalf* enW2t = aB((size_t)512 * 512);
  bhalf* enW3t = aB((size_t)256 * 512);
  bhalf* eeW1t = aB((size_t)512 * 32);
  bhalf* eeW2t = aB((size_t)512 * 512);
  bhalf* eeW3t = aB((size_t)256 * 512);
  bhalf* eW1t  = aB((size_t)LLAYER * 512 * 1024);
  bhalf* eW2t  = aB((size_t)LLAYER * 512 * 512);
  bhalf* eW3t  = aB((size_t)LLAYER * 256 * 512);
  bhalf* nW1t  = aB((size_t)LLAYER * 512 * 1024);
  bhalf* nW2t  = aB((size_t)LLAYER * 512 * 512);
  bhalf* nW3t  = aB((size_t)LLAYER * 256 * 512);
  bhalf* gW1t  = aB((size_t)LLAYER * 512 * 768);
  bhalf* gW2t  = aB((size_t)LLAYER * 512 * 512);
  bhalf* gW3t  = aB((size_t)LLAYER * 256 * 512);
  bhalf* pW1t  = aB((size_t)LLAYER * 256 * 256);
  bhalf* pW2t  = aB((size_t)LLAYER * 3 * 256);
  // activations
  bhalf* xrb = aB((size_t)NNODE * 192);
  bhalf* erb = aB((size_t)NEDGE * 32);
  bhalf* xB  = aB((size_t)NNODE * DD);
  bhalf* eB  = aB((size_t)NEDGE * DD);
  bhalf* uB  = aB((size_t)NGRAPH * DD);
  bhalf* eIn = aB((size_t)NEDGE * 4 * DD);
  bhalf* nIn = aB((size_t)NNODE * 4 * DD);
  bhalf* gIn = aB((size_t)NGRAPH * 3 * DD);
  bhalf* h1  = aB((size_t)NEDGE * HH);
  bhalf* h2  = aB((size_t)NEDGE * HH);
  bhalf* ph  = aB((size_t)NNODE * DD);
  float* xF    = aF((size_t)NNODE * DD);
  float* eF    = aF((size_t)NEDGE * DD);
  float* uF    = aF((size_t)NGRAPH * DD);
  float* sentF = aF((size_t)NNODE * DD);
  float* recvF = aF((size_t)NNODE * DD);
  float* gnF   = aF((size_t)NGRAPH * DD);
  float* geF   = aF((size_t)NGRAPH * DD);

  auto packW = [&](const float* W, bhalf* Wt, int K, int Nc, int Kp) {
    const long tot = (long)Nc * Kp;
    pack_wt_kernel<<<dim3((unsigned)((tot + 255) / 256)), dim3(256), 0, stream>>>(W, Wt, K, Nc, Kp);
  };
  auto zero = [&](float* p, long n) {
    zero_f32_kernel<<<dim3(2048), dim3(256), 0, stream>>>(p, n);
  };
  auto gemm = [&](int mode, const bhalf* A, const bhalf* Wt, const float* bias,
                  bhalf* oB, float* oF, int M, int Nc, int K) {
    dim3 g((unsigned)((M + BM - 1) / BM), (unsigned)((Nc + BN - 1) / BN));
    dim3 b(256);
    switch (mode) {
      case 0:  gemm_bf16_kernel<0><<<g, b, 0, stream>>>(A, Wt, bias, oB, oF, M, Nc, K); break;
      case 1:  gemm_bf16_kernel<1><<<g, b, 0, stream>>>(A, Wt, bias, oB, oF, M, Nc, K); break;
      case 2:  gemm_bf16_kernel<2><<<g, b, 0, stream>>>(A, Wt, bias, oB, oF, M, Nc, K); break;
      default: gemm_bf16_kernel<3><<<g, b, 0, stream>>>(A, Wt, bias, oB, oF, M, Nc, K); break;
    }
  };

  // --- weight packing (one pass; cheap relative to ~6 TFLOP of GEMM) ---
  packW(enW1, enW1t, 173, 512, 192);
  packW(enW2, enW2t, 512, 512, 512);
  packW(enW3, enW3t, 512, 256, 512);
  packW(eeW1, eeW1t, 13, 512, 32);
  packW(eeW2, eeW2t, 512, 512, 512);
  packW(eeW3, eeW3t, 512, 256, 512);
  for (int l = 0; l < LLAYER; ++l) {
    packW(eW1 + (size_t)l * 1024 * 512, eW1t + (size_t)l * 512 * 1024, 1024, 512, 1024);
    packW(eW2 + (size_t)l * 512 * 512,  eW2t + (size_t)l * 512 * 512,  512, 512, 512);
    packW(eW3 + (size_t)l * 512 * 256,  eW3t + (size_t)l * 256 * 512,  512, 256, 512);
    packW(nW1 + (size_t)l * 1024 * 512, nW1t + (size_t)l * 512 * 1024, 1024, 512, 1024);
    packW(nW2 + (size_t)l * 512 * 512,  nW2t + (size_t)l * 512 * 512,  512, 512, 512);
    packW(nW3 + (size_t)l * 512 * 256,  nW3t + (size_t)l * 256 * 512,  512, 256, 512);
    packW(gW1 + (size_t)l * 768 * 512,  gW1t + (size_t)l * 512 * 768,  768, 512, 768);
    packW(gW2 + (size_t)l * 512 * 512,  gW2t + (size_t)l * 512 * 512,  512, 512, 512);
    packW(gW3 + (size_t)l * 512 * 256,  gW3t + (size_t)l * 256 * 512,  512, 256, 512);
    packW(pW1 + (size_t)l * 256 * 256,  pW1t + (size_t)l * 256 * 256,  256, 256, 256);
    packW(pW2 + (size_t)l * 256 * 3,    pW2t + (size_t)l * 3 * 256,    256, 3,   256);
  }

  // --- input packing + state init ---
  {
    long tot = (long)NNODE * 192;
    pack_rows_kernel<<<dim3((unsigned)((tot + 255) / 256)), dim3(256), 0, stream>>>(x_raw, xrb, NNODE, 173, 192);
    tot = (long)NEDGE * 32;
    pack_rows_kernel<<<dim3((unsigned)((tot + 255) / 256)), dim3(256), 0, stream>>>(edge_raw, erb, NEDGE, 13, 32);
  }
  init_u_kernel<<<dim3(NGRAPH), dim3(DD), 0, stream>>>(u0, uF, uB);
  zero(pos, (long)NNODE * 3);

  // --- encoders ---
  gemm(0, xrb, enW1t, enb1, h1, nullptr, NNODE, HH, 192);
  gemm(0, h1,  enW2t, enb2, h2, nullptr, NNODE, HH, HH);
  gemm(1, h2,  enW3t, enb3, xB, xF,      NNODE, DD, HH);
  gemm(0, erb, eeW1t, eeb1, h1, nullptr, NEDGE, HH, 32);
  gemm(0, h1,  eeW2t, eeb2, h2, nullptr, NEDGE, HH, HH);
  gemm(1, h2,  eeW3t, eeb3, eB, eF,      NEDGE, DD, HH);

  // --- 12 message-passing layers ---
  for (int l = 0; l < LLAYER; ++l) {
    // edge update: e += relu-MLP(concat(e, x[src], x[dst], u[batch]))
    build_e_in_kernel<<<dim3(NEDGE), dim3(128), 0, stream>>>(eB, xB, uB, srcI, dstI, node_batch, eIn);
    gemm(0, eIn, eW1t + (size_t)l * 512 * 1024, eb1 + (size_t)l * HH, h1, nullptr, NEDGE, HH, 4 * DD);
    gemm(0, h1,  eW2t + (size_t)l * 512 * 512,  eb2 + (size_t)l * HH, h2, nullptr, NEDGE, HH, HH);
    gemm(2, h2,  eW3t + (size_t)l * 256 * 512,  eb3 + (size_t)l * DD, eB, eF,      NEDGE, DD, HH);

    // node update: x += relu-MLP(concat(x, sent, recv, u[batch]))
    zero(sentF, (long)NNODE * DD);
    zero(recvF, (long)NNODE * DD);
    scatter_e_kernel<<<dim3(NEDGE), dim3(DD), 0, stream>>>(eF, srcI, dstI, sentF, recvF);
    build_n_in_kernel<<<dim3(NNODE), dim3(128), 0, stream>>>(xB, sentF, recvF, uB, node_batch, nIn);
    gemm(0, nIn, nW1t + (size_t)l * 512 * 1024, nb1 + (size_t)l * HH, h1, nullptr, NNODE, HH, 4 * DD);
    gemm(0, h1,  nW2t + (size_t)l * 512 * 512,  nb2 + (size_t)l * HH, h2, nullptr, NNODE, HH, HH);
    gemm(2, h2,  nW3t + (size_t)l * 256 * 512,  nb3 + (size_t)l * DD, xB, xF,      NNODE, DD, HH);

    // global update: u += relu-MLP(concat(u, sum_x, sum_e))
    zero(gnF, (long)NGRAPH * DD);
    zero(geF, (long)NGRAPH * DD);
    scatter_gn_kernel<<<dim3(NNODE), dim3(DD), 0, stream>>>(xF, node_batch, gnF);
    scatter_ge_kernel<<<dim3(NEDGE), dim3(DD), 0, stream>>>(eF, srcI, node_batch, geF);
    build_g_in_kernel<<<dim3(NGRAPH), dim3(96), 0, stream>>>(uB, gnF, geF, gIn);
    gemm(0, gIn, gW1t + (size_t)l * 512 * 768, gb1 + (size_t)l * HH, h1, nullptr, NGRAPH, HH, 3 * DD);
    gemm(0, h1,  gW2t + (size_t)l * 512 * 512, gb2 + (size_t)l * HH, h2, nullptr, NGRAPH, HH, HH);
    gemm(2, h2,  gW3t + (size_t)l * 256 * 512, gb3 + (size_t)l * DD, uB, uF,      NGRAPH, DD, HH);

    // position head: pos += relu(x @ pW1 + pb1) @ pW2 + pb2
    gemm(0, xB, pW1t + (size_t)l * 256 * 256, pb1 + (size_t)l * DD, ph, nullptr, NNODE, DD, DD);
    gemm(3, ph, pW2t + (size_t)l * 3 * 256,   pb2 + (size_t)l * 3,  nullptr, pos, NNODE, 3, DD);
  }
}
